// BarrierPolicy_2439541424408
// MI455X (gfx1250) — matrix operand order repaired and checked by reading, not compiled, not run on hardware
//
#include <hip/hip_runtime.h>

typedef __attribute__((ext_vector_type(16))) _Float16 v16h;
typedef __attribute__((ext_vector_type(8)))  float    v8f;
typedef __attribute__((ext_vector_type(4)))  float    v4f;

#define NDIM 8
#define MDIM 8
#define H1D  64
#define R2C  16.0f
#define UMINC (-1.0f)
#define UMAXC (1.0f)
#define ROWS_PER_WG 128
#define X2_STRIDE 65   // floats; 65 mod 64 banks -> conflict-free per-lane rows
#define H1_STRIDE 72   // halves; 144B rows, 16B aligned chunks

__device__ __forceinline__ void lds_wave_fence() {
  asm volatile("s_wait_dscnt 0x0" ::: "memory");
}

__global__ __launch_bounds__(128) void cbf_qp_policy_kernel(
    const float* __restrict__ x,   const float* __restrict__ W1,  const float* __restrict__ b1,
    const float* __restrict__ W21, const float* __restrict__ b21,
    const float* __restrict__ W22, const float* __restrict__ b22,
    const float* __restrict__ W31, const float* __restrict__ b31,
    const float* __restrict__ W32, const float* __restrict__ b32,
    const float* __restrict__ Amat, const float* __restrict__ Gmat,
    const float* __restrict__ meanv, const float* __restrict__ stdv,
    float* __restrict__ out, int Btot)
{
  __shared__ _Float16 h1s[4][16 * H1_STRIDE];          // per-wave 16x64 f16 stage
  __shared__ float    x2s[4][32 * X2_STRIDE];          // per-wave 32x64 f32 stage

  const int tid  = threadIdx.x;
  const int wave = tid >> 5;
  const int lane = tid & 31;
  const int l16  = lane & 15;
  const int hi16 = lane >> 4;                          // 0 or 1
  const int wave_row0 = blockIdx.x * ROWS_PER_WG + wave * 32;

  // ---------------- B fragments (hoisted, register-resident) ----------------
  // Layer 1: W1 is (8,64) row-major; K padded 8->32. B-frag: lane -> n=16j+l16,
  // halves h -> k = 16*hi16 + h (valid only k<8 on lanes 0-15).
  v16h b1f[4];
  #pragma unroll
  for (int j = 0; j < 4; ++j) {
    v16h f = {};
    if (hi16 == 0) {
      int n = 16 * j + l16;
      #pragma unroll
      for (int k = 0; k < 8; ++k) f[k] = (_Float16)W1[k * 64 + n];
    }
    b1f[j] = f;
  }
  // Layer 2: W2 = [W21 | W22] (64x64). 4 col-tiles x 2 K-chunks of 32.
  v16h b2f[4][2];
  #pragma unroll
  for (int j = 0; j < 4; ++j) {
    const float* W = (j < 2) ? W21 : W22;
    const int n  = 16 * j + l16;
    const int nl = (j < 2) ? n : (n - 32);
    #pragma unroll
    for (int kc = 0; kc < 2; ++kc) {
      v16h f;
      #pragma unroll
      for (int h = 0; h < 16; ++h) {
        int k = 32 * kc + 16 * hi16 + h;
        f[h] = (_Float16)W[k * 32 + nl];
      }
      b2f[j][kc] = f;
    }
  }
  float bias1[4], bias2[4];
  #pragma unroll
  for (int j = 0; j < 4; ++j) {
    int n = 16 * j + l16;
    bias1[j] = b1[n];
    bias2[j] = (j < 2) ? b21[n] : b22[n - 32];
  }

  // ---------------- Two 16-row WMMA tiles per wave ----------------
  #pragma unroll
  for (int t = 0; t < 2; ++t) {
    const int r0 = wave_row0 + 16 * t;
    // A1 from x: lanes 0-15, halves 0-7 = x[r0+l16][0..7]; rest zero (K pad).
    v16h a1 = {};
    if (hi16 == 0) {
      int rr = r0 + l16; if (rr >= Btot) rr = Btot - 1;
      const v4f* xv = (const v4f*)(x + (size_t)rr * NDIM);
      v4f x0v = xv[0], x1v = xv[1];
      #pragma unroll
      for (int k = 0; k < 4; ++k) { a1[k] = (_Float16)x0v[k]; a1[4 + k] = (_Float16)x1v[k]; }
    }
    // Layer 1: 4 WMMAs -> h1 C-tiles; bias+ReLU; stage f16 to LDS.
    #pragma unroll
    for (int j = 0; j < 4; ++j) {
      v8f acc = {};
      acc = __builtin_amdgcn_wmma_f32_16x16x32_f16(false, a1, false, b1f[j],
                                                   (short)0, acc, false, false);
      const int n = 16 * j + l16;
      #pragma unroll
      for (int v = 0; v < 8; ++v) {
        float e = fmaxf(acc[v] + bias1[j], 0.0f);
        h1s[wave][(v + 8 * hi16) * H1_STRIDE + n] = (_Float16)e;
      }
    }
    lds_wave_fence();
    // Layer 2 A-fragments from LDS (contiguous k = 16*hi16 + h, matching B-frags).
    v16h a2[2];
    #pragma unroll
    for (int kc = 0; kc < 2; ++kc) {
      const int c0 = 32 * kc + 16 * hi16;
      const _Float16* row = &h1s[wave][l16 * H1_STRIDE];
      v16h f;
      #pragma unroll
      for (int h = 0; h < 8; ++h) { f[h] = row[c0 + h]; f[8 + h] = row[c0 + 8 + h]; }
      a2[kc] = f;
    }
    // Layer 2: 8 WMMAs (K=64 chained) -> x2 tiles; bias+ReLU; stage f32 to LDS.
    #pragma unroll
    for (int j = 0; j < 4; ++j) {
      v8f acc = {};
      acc = __builtin_amdgcn_wmma_f32_16x16x32_f16(false, a2[0], false, b2f[j][0],
                                                   (short)0, acc, false, false);
      acc = __builtin_amdgcn_wmma_f32_16x16x32_f16(false, a2[1], false, b2f[j][1],
                                                   (short)0, acc, false, false);
      const int n = 16 * j + l16;
      #pragma unroll
      for (int v = 0; v < 8; ++v) {
        float e = fmaxf(acc[v] + bias2[j], 0.0f);
        x2s[wave][(16 * t + v + 8 * hi16) * X2_STRIDE + n] = e;
      }
    }
  }
  lds_wave_fence();

  // ---------------- Per-lane: layer 3, barrier terms, exact QP ----------------
  int gr = wave_row0 + lane;
  const bool live = (gr < Btot);
  if (gr >= Btot) gr = Btot - 1;
  const float* x2r = &x2s[wave][lane * X2_STRIDE];     // [x21(32) | x22(32)]

  float px[MDIM];
  #pragma unroll
  for (int m = 0; m < MDIM; ++m) px[m] = b31[m];
  #pragma unroll 4
  for (int k = 0; k < 32; ++k) {
    const float xk = x2r[k];
    #pragma unroll
    for (int m = 0; m < MDIM; ++m) px[m] = fmaf(xk, W31[k * MDIM + m], px[m]);
  }
  float s = b32[0];
  #pragma unroll 8
  for (int k = 0; k < 32; ++k) s = fmaf(x2r[32 + k], W32[k], s);
  const float alpha = 4.0f / (1.0f + expf(-s));

  float x0[NDIM], dh[NDIM];
  float hx = R2C;
  {
    const float* xr = x + (size_t)gr * NDIM;
    #pragma unroll
    for (int i = 0; i < NDIM; ++i) {
      float v = fmaf(xr[i], stdv[i], meanv[i]);
      x0[i] = v; dh[i] = -2.0f * v;
      hx = fmaf(-v, v, hx);
    }
  }
  float Lfhx = 0.0f;
  #pragma unroll
  for (int i = 0; i < NDIM; ++i) {
    float fi = 0.0f;
    #pragma unroll
    for (int j = 0; j < NDIM; ++j) fi = fmaf(Amat[i * NDIM + j], x0[j], fi);
    Lfhx = fmaf(dh[i], fi, Lfhx);
  }
  float g[MDIM];
  #pragma unroll
  for (int m = 0; m < MDIM; ++m) {
    float gm = 0.0f;
    #pragma unroll
    for (int i = 0; i < NDIM; ++i) gm = fmaf(dh[i], Gmat[i * MDIM + m], gm);
    g[m] = gm;
  }
  const float c0 = Lfhx + alpha * hx;

  auto c_of = [&](float lam) -> float {
    float c = c0;
    #pragma unroll
    for (int m = 0; m < MDIM; ++m) {
      float u = fminf(fmaxf(fmaf(lam, g[m], -px[m]), UMINC), UMAXC);
      c = fmaf(g[m], u, c);
    }
    return c;
  };

  // bracket lambda* in [0, hiv]
  float hiv = 1.0f;
  #pragma unroll 1
  for (int i = 0; i < 40; ++i) hiv = (c_of(hiv) < 0.0f) ? 2.0f * hiv : hiv;
  // bisection
  float lov = 0.0f;
  #pragma unroll 1
  for (int i = 0; i < 60; ++i) {
    const float mid = 0.5f * (lov + hiv);
    const bool neg = (c_of(mid) < 0.0f);
    lov = neg ? mid : lov;
    hiv = neg ? hiv : mid;
  }
  const float lam_b = 0.5f * (lov + hiv);

  // frozen active set -> closed-form KKT-consistent lambda
  float denom = 0.0f, num = c0;
  bool lowm[MDIM], highm[MDIM];
  #pragma unroll
  for (int m = 0; m < MDIM; ++m) {
    const float ur = fmaf(lam_b, g[m], -px[m]);
    const bool lo_ = (ur <= UMINC), hi_ = (ur >= UMAXC);
    lowm[m] = lo_; highm[m] = hi_;
    const bool fr = !(lo_ || hi_);
    denom += fr ? g[m] * g[m] : 0.0f;
    num   += lo_ ? UMINC * g[m] : 0.0f;
    num   += hi_ ? UMAXC * g[m] : 0.0f;
    num   -= fr  ? g[m] * px[m] : 0.0f;
  }
  const bool viol = (c_of(0.0f) < 0.0f);
  const float lam = viol ? (-num / (denom + 1e-12f)) : 0.0f;

  v8f u;
  #pragma unroll
  for (int m = 0; m < MDIM; ++m) {
    const float uf = fmaf(lam, g[m], -px[m]);
    u[m] = lowm[m] ? UMINC : (highm[m] ? UMAXC : uf);
  }
  if (live) {
    v4f* o = (v4f*)(out + (size_t)gr * MDIM);
    o[0] = v4f{u[0], u[1], u[2], u[3]};
    o[1] = v4f{u[4], u[5], u[6], u[7]};
  }
}

extern "C" void kernel_launch(void* const* d_in, const int* in_sizes, int n_in,
                              void* d_out, int out_size, void* d_ws, size_t ws_size,
                              hipStream_t stream) {
  (void)n_in; (void)d_ws; (void)ws_size;
  const float* x    = (const float*)d_in[0];
  const float* W1   = (const float*)d_in[1];
  const float* b1   = (const float*)d_in[2];
  const float* W21  = (const float*)d_in[3];
  const float* b21  = (const float*)d_in[4];
  const float* W22  = (const float*)d_in[5];
  const float* b22  = (const float*)d_in[6];
  const float* W31  = (const float*)d_in[7];
  const float* b31  = (const float*)d_in[8];
  const float* W32  = (const float*)d_in[9];
  const float* b32  = (const float*)d_in[10];
  const float* A    = (const float*)d_in[11];
  const float* G    = (const float*)d_in[12];
  const float* mean = (const float*)d_in[13];
  const float* stdv = (const float*)d_in[14];

  const int Btot   = in_sizes[0] / NDIM;               // 262144
  const int blocks = (Btot + ROWS_PER_WG - 1) / ROWS_PER_WG;
  cbf_qp_policy_kernel<<<blocks, 128, 0, stream>>>(
      x, W1, b1, W21, b21, W22, b22, W31, b31, W32, b32, A, G, mean, stdv,
      (float*)d_out, Btot);
}
